// DGDAGRNN_80350248173911
// MI455X (gfx1250) — compile-verified, matrix operand order checked by reading, not run on hardware
//
#include <hip/hip_runtime.h>
#include <hip/hip_bf16.h>

#define VHS 100
#define NVT 6
#define CHS 30
#define NROUNDS 3
#define P_ 8192
#define L_ 12
#define DEG_ 4
#define N_ (P_ * L_)
#define S_ 2048
#define NCL 4

typedef __attribute__((ext_vector_type(16))) _Float16 v16h;
typedef __attribute__((ext_vector_type(8)))  _Float16 v8h;
typedef __attribute__((ext_vector_type(8)))  float    v8f;

__device__ __forceinline__ float sigf(float x) { return 1.0f / (1.0f + __expf(-x)); }

// ---------------------------------------------------------------------------
// Generic WMMA GEMM: C[M x Ntiles*16] f32 = A(M x Kp f16 row-major) * B^T
// where B is stored as (Np x Kp) f16 row-major (i.e. weight layout [out][in]).
// One wave computes one 16x16 tile. act: 0=none, 1=sigmoid, 2=relu.
// ---------------------------------------------------------------------------
__global__ __launch_bounds__(32) void gemm_wmma(
    const _Float16* __restrict__ A, const _Float16* __restrict__ B,
    const float* __restrict__ bias, float* __restrict__ C,
    int Kp, int ldc, int Nvalid, int act)
{
    const int mt   = blockIdx.x;
    const int nt   = blockIdx.y;
    const int lane = threadIdx.x;
    const int half = lane >> 4;   // 0 or 1
    const int rl   = lane & 15;

    const _Float16* ap = A + (size_t)(mt * 16 + rl) * Kp + half * 8;
    const _Float16* bp = B + (size_t)(nt * 16 + rl) * Kp + half * 8;

    v8f acc = {};
    for (int kc = 0; kc < Kp; kc += 32) {
        v8h alo = *(const v8h*)(ap + kc);
        v8h ahi = *(const v8h*)(ap + kc + 16);
        v8h blo = *(const v8h*)(bp + kc);
        v8h bhi = *(const v8h*)(bp + kc + 16);
        v16h av = __builtin_shufflevector(alo, ahi, 0,1,2,3,4,5,6,7,8,9,10,11,12,13,14,15);
        v16h bv = __builtin_shufflevector(blo, bhi, 0,1,2,3,4,5,6,7,8,9,10,11,12,13,14,15);
        acc = __builtin_amdgcn_wmma_f32_16x16x32_f16(false, av, false, bv,
                                                     (short)0, acc, false, false);
    }

    const int col = nt * 16 + rl;
    const float bval = (bias && col < Nvalid) ? bias[col] : 0.0f;
#pragma unroll
    for (int v = 0; v < 8; v++) {
        int row = mt * 16 + v + 8 * half;
        float x = acc[v] + bval;
        if (act == 1)      x = sigf(x);
        else if (act == 2) x = fmaxf(x, 0.0f);
        C[(size_t)row * ldc + col] = x;
    }
}

// ---------------------------------------------------------------------------
// Pack / gather f32 -> padded f16 (rows Rp x Kp), zero-filled padding.
// Row source: base + (idx ? idx[r] : r), source leading dim lds.
// ---------------------------------------------------------------------------
__global__ void pack_f16(const float* __restrict__ src, int lds,
                         const int* __restrict__ idx, int base,
                         int rows, int K, int Kp, int Rp,
                         _Float16* __restrict__ dst)
{
    size_t i   = (size_t)blockIdx.x * blockDim.x + threadIdx.x;
    size_t tot = (size_t)Rp * Kp;
    if (i >= tot) return;
    int r = (int)(i / Kp);
    int k = (int)(i % Kp);
    float v = 0.0f;
    if (r < rows && k < K) {
        int sr = base + (idx ? idx[r] : r);
        v = src[(size_t)sr * lds + k];
    }
    dst[i] = (_Float16)v;
}

// msg = sigmoid(gate) * mapv, summed over the 4 contiguous edges per target.
__global__ void msg_segsum(const float* __restrict__ g, const float* __restrict__ m,
                           float* __restrict__ ps)
{
    int t = blockIdx.x;
    int c = threadIdx.x;
    if (c >= 112) return;
    float acc = 0.0f;
#pragma unroll
    for (int j = 0; j < 4; j++) {
        size_t e = (size_t)(t * 4 + j) * 112 + c;
        acc += g[e] * m[e];
    }
    ps[(size_t)t * 112 + c] = acc;
}

// GRU gating. gi stride gis (0 = broadcast single row), gh stride fixed 304.
__global__ void gru_pointwise(const float* __restrict__ gi, int gis,
                              const float* __restrict__ gh,
                              const float* __restrict__ hx, int ldhx,
                              const int* __restrict__ hxidx, int hxbase,
                              float* __restrict__ out, int ldo,
                              const int* __restrict__ oidx, int obase)
{
    int r = blockIdx.x;
    int c = threadIdx.x;
    if (c >= VHS) return;
    size_t gio = (size_t)r * gis;
    size_t gho = (size_t)r * 304;
    float ir = gi[gio + c],       iz = gi[gio + 100 + c], in_ = gi[gio + 200 + c];
    float hr = gh[gho + c],       hz = gh[gho + 100 + c], hn  = gh[gho + 200 + c];
    int hrow = hxbase + (hxidx ? hxidx[r] : r);
    float hv = hx[(size_t)hrow * ldhx + c];
    float rg = sigf(ir + hr);
    float z  = sigf(iz + hz);
    float n  = tanhf(in_ + rg * hn);
    float o  = (1.0f - z) * n + z * hv;
    int orow = obase + (oidx ? oidx[r] : r);
    out[(size_t)orow * ldo + c] = o;
}

// gi row for output-node GRU: girow = bih + wih @ h[onode]
__global__ void outnode_gi(const float* __restrict__ wih, const float* __restrict__ bih,
                           const float* __restrict__ h, const int* __restrict__ onode,
                           float* __restrict__ girow)
{
    int j = blockIdx.x * blockDim.x + threadIdx.x;
    if (j >= 300) return;
    int row = onode[0];
    float acc = bih[j];
    const float* wr = wih + (size_t)j * VHS;
    const float* hr = h + (size_t)row * VHS;
    for (int k = 0; k < VHS; k++) acc += wr[k] * hr[k];
    girow[j] = acc;
}

// Serial LSTM recurrence, batch 1, 2048 steps. blockIdx.x = direction (0 fwd, 1 bwd).
// g = XW[t] + bhh + Whh @ h ; outputs h_t into outH[t*ldo + dir*100 + j].
__global__ __launch_bounds__(512) void lstm_recur(
    const float* __restrict__ XWf, const float* __restrict__ XWb,
    const float* __restrict__ whh_f, const float* __restrict__ whh_b,
    const float* __restrict__ bhh_f, const float* __restrict__ bhh_b,
    float* __restrict__ outH, int ldo)
{
    __shared__ float h_s[VHS];
    __shared__ float g_s[4 * VHS];
    const int dir = blockIdx.x;
    const float* XW  = dir ? XWb  : XWf;
    const float* whh = dir ? whh_b : whh_f;
    const float* bhh = dir ? bhh_b : bhh_f;
    const int j = threadIdx.x;
    float cst = 0.0f;
    if (j < VHS) h_s[j] = 0.0f;
    __syncthreads();
    for (int s = 0; s < S_; s++) {
        int t = dir ? (S_ - 1 - s) : s;
        if (j < 4 * VHS) {
            float acc = XW[(size_t)t * 400 + j] + bhh[j];
            const float* wr = whh + (size_t)j * VHS;
#pragma unroll 4
            for (int k = 0; k < VHS; k++) acc += wr[k] * h_s[k];
            g_s[j] = acc;
        }
        __syncthreads();
        if (j < VHS) {
            float ig = sigf(g_s[j]);
            float fg = sigf(g_s[100 + j]);
            float gg = tanhf(g_s[200 + j]);
            float og = sigf(g_s[300 + j]);
            cst = fg * cst + ig * gg;
            float hv = og * tanhf(cst);
            h_s[j] = hv;
            outH[(size_t)t * ldo + dir * VHS + j] = hv;
        }
        __syncthreads();
    }
}

// cg = tanh(relu_t1 @ m2^T + m2_b), one value per sequence position.
__global__ void final_out(const float* __restrict__ t1, const float* __restrict__ m2w,
                          const float* __restrict__ m2b, float* __restrict__ out)
{
    int s = blockIdx.x * blockDim.x + threadIdx.x;
    if (s >= S_) return;
    float acc = m2b[0];
    for (int j = 0; j < CHS; j++) acc += t1[(size_t)s * 32 + j] * m2w[j];
    out[s] = tanhf(acc);
}

__global__ void h_init(const float* __restrict__ w_init, float* __restrict__ h)
{
    size_t i = (size_t)blockIdx.x * blockDim.x + threadIdx.x;
    if (i >= (size_t)N_ * VHS) return;
    h[i] = w_init[i % VHS];
}

// ---------------------------------------------------------------------------
extern "C" void kernel_launch(void* const* d_in, const int* in_sizes, int n_in,
                              void* d_out, int out_size, void* d_ws, size_t ws_size,
                              hipStream_t stream)
{
    (void)in_sizes; (void)n_in; (void)out_size; (void)ws_size;
    const float* x        = (const float*)d_in[0];
    const float* w_init   = (const float*)d_in[1];
    const float* gi_wih   = (const float*)d_in[2];
    const float* gi_whh   = (const float*)d_in[3];
    const float* gi_bih   = (const float*)d_in[4];
    const float* gi_bhh   = (const float*)d_in[5];
    const float* gh_wih   = (const float*)d_in[6];
    const float* gh_whh   = (const float*)d_in[7];
    const float* gh_bih   = (const float*)d_in[8];
    const float* gh_bhh   = (const float*)d_in[9];
    const float* gl_wih   = (const float*)d_in[10];
    const float* gl_whh   = (const float*)d_in[11];
    const float* gl_bih   = (const float*)d_in[12];
    const float* gl_bhh   = (const float*)d_in[13];
    const float* go_wih   = (const float*)d_in[14];
    const float* go_whh   = (const float*)d_in[15];
    const float* go_bih   = (const float*)d_in[16];
    const float* go_bhh   = (const float*)d_in[17];
    const float* gate_w   = (const float*)d_in[18];
    const float* gate_b   = (const float*)d_in[19];
    const float* map_w    = (const float*)d_in[20];
    const float* lu_wih_f = (const float*)d_in[21];
    const float* lu_whh_f = (const float*)d_in[22];
    const float* lu_bih_f = (const float*)d_in[23];
    const float* lu_bhh_f = (const float*)d_in[24];
    const float* lu_wih_b = (const float*)d_in[25];
    const float* lu_whh_b = (const float*)d_in[26];
    const float* lu_bih_b = (const float*)d_in[27];
    const float* lu_bhh_b = (const float*)d_in[28];
    const float* lg_wih_f = (const float*)d_in[29];
    const float* lg_whh_f = (const float*)d_in[30];
    const float* lg_bih_f = (const float*)d_in[31];
    const float* lg_bhh_f = (const float*)d_in[32];
    const float* lg_wih_b = (const float*)d_in[33];
    const float* lg_whh_b = (const float*)d_in[34];
    const float* lg_bih_b = (const float*)d_in[35];
    const float* lg_bhh_b = (const float*)d_in[36];
    const float* u21_w    = (const float*)d_in[37];
    const float* u21_b    = (const float*)d_in[38];
    const float* m1_w     = (const float*)d_in[39];
    const float* m1_b     = (const float*)d_in[40];
    const float* m2_w     = (const float*)d_in[41];
    const float* m2_b     = (const float*)d_in[42];
    const int* edge_index = (const int*)d_in[43];
    const int* sv_node    = (const int*)d_in[44];
    const int* li_node    = (const int*)d_in[45];
    const int* out_node   = (const int*)d_in[46];
    float* out = (float*)d_out;

    // ---- workspace bump allocator (256B aligned) ----
    size_t off = 0;
    auto alloc = [&](size_t bytes) -> void* {
        off = (off + 255) & ~(size_t)255;
        void* p = (char*)d_ws + off;
        off += bytes;
        return p;
    };
    float*    h    = (float*)alloc((size_t)N_ * VHS * 4);        // node states
    _Float16* A16  = (_Float16*)alloc((size_t)32768 * 224 * 2);  // f16 A-operand scratch
    float*    F1   = (float*)alloc((size_t)32768 * 112 * 4);     // gemm out / gi / XWf / t1
    float*    F2   = (float*)alloc((size_t)32768 * 112 * 4);     // gemm out / gh / XWb
    float*    PS   = (float*)alloc((size_t)P_ * 112 * 4);        // aggregated msgs / pred
    float*    ABUF = (float*)alloc((size_t)S_ * 112 * 4);        // 'a' GRU output
    float*    PROP = (float*)alloc((size_t)S_ * 200 * 4);        // bilstm prop
    float*    SVF  = (float*)alloc((size_t)S_ * 112 * 4);        // sv_feature (clauses >= 1)
    float*    GIROW= (float*)alloc(304 * 4);
    // packed f16 weights (row-major [out][Kp], padded with zeros)
    _Float16* Wgate   = (_Float16*)alloc(112 * 128 * 2);
    _Float16* Wmap    = (_Float16*)alloc(112 * 128 * 2);
    _Float16* Wgiw    = (_Float16*)alloc(304 * 32 * 2);
    _Float16* Wgihh   = (_Float16*)alloc(304 * 128 * 2);
    _Float16* Wghw    = (_Float16*)alloc(304 * 128 * 2);
    _Float16* Wghhh   = (_Float16*)alloc(304 * 128 * 2);
    _Float16* Wglw    = (_Float16*)alloc(304 * 128 * 2);
    _Float16* Wglhh   = (_Float16*)alloc(304 * 128 * 2);
    _Float16* Wgow    = (_Float16*)alloc(304 * 128 * 2);   // (unused in gemm; out-node matvec uses f32)
    _Float16* Wgohh   = (_Float16*)alloc(304 * 128 * 2);
    _Float16* Wluf    = (_Float16*)alloc(400 * 128 * 2);
    _Float16* Wlub    = (_Float16*)alloc(400 * 128 * 2);
    _Float16* Wlgf    = (_Float16*)alloc(400 * 224 * 2);
    _Float16* Wlgb    = (_Float16*)alloc(400 * 224 * 2);
    _Float16* Wu21    = (_Float16*)alloc(112 * 224 * 2);
    _Float16* Wm1     = (_Float16*)alloc(32 * 224 * 2);

    auto packW = [&](const float* src, int rows, int K, int Kp, int Rp, _Float16* dst) {
        size_t tot = (size_t)Rp * Kp;
        pack_f16<<<(int)((tot + 255) / 256), 256, 0, stream>>>(src, K, nullptr, 0, rows, K, Kp, Rp, dst);
    };
    auto packA = [&](const float* src, int lds, const int* idx, int base,
                     int rows, int K, int Kp, _Float16* dst) {
        size_t tot = (size_t)rows * Kp;
        pack_f16<<<(int)((tot + 255) / 256), 256, 0, stream>>>(src, lds, idx, base, rows, K, Kp, rows, dst);
    };
    auto gemm = [&](const _Float16* A, const _Float16* B, const float* bias, float* C,
                    int Mrows, int Ntiles, int Kp, int Nvalid, int act) {
        dim3 g(Mrows / 16, Ntiles);
        gemm_wmma<<<g, 32, 0, stream>>>(A, B, bias, C, Kp, Ntiles * 16, Nvalid, act);
    };

    // ---- pack weights ----
    packW(gate_w, 100, 100, 128, 112, Wgate);
    packW(map_w,  100, 100, 128, 112, Wmap);
    packW(gi_wih, 300,   6,  32, 304, Wgiw);
    packW(gi_whh, 300, 100, 128, 304, Wgihh);
    packW(gh_wih, 300, 100, 128, 304, Wghw);
    packW(gh_whh, 300, 100, 128, 304, Wghhh);
    packW(gl_wih, 300, 100, 128, 304, Wglw);
    packW(gl_whh, 300, 100, 128, 304, Wglhh);
    packW(go_wih, 300, 100, 128, 304, Wgow);
    packW(go_whh, 300, 100, 128, 304, Wgohh);
    packW(lu_wih_f, 400, 100, 128, 400, Wluf);
    packW(lu_wih_b, 400, 100, 128, 400, Wlub);
    packW(lg_wih_f, 400, 200, 224, 400, Wlgf);
    packW(lg_wih_b, 400, 200, 224, 400, Wlgb);
    packW(u21_w, 100, 200, 224, 112, Wu21);
    packW(m1_w,   30, 200, 224,  32, Wm1);

    h_init<<<(N_ * VHS + 255) / 256, 256, 0, stream>>>(w_init, h);

    const int NE = P_ * DEG_;  // 32768 edges per layer
    for (int r = 0; r < NROUNDS; r++) {
        for (int l = 1; l < L_; l++) {
            const int* src_idx = edge_index + (size_t)(l - 1) * NE;  // row 0 slice
            // gather sources, gate/map matmuls, fused sigmoid
            packA(h, VHS, src_idx, 0, NE, VHS, 128, A16);
            gemm(A16, Wgate, gate_b, F1, NE, 7, 128, VHS, 1);
            gemm(A16, Wmap, nullptr, F2, NE, 7, 128, VHS, 0);
            msg_segsum<<<P_, 128, 0, stream>>>(F1, F2, PS);
            // GRU input projection
            if (r == 0) {
                packA(x, NVT, nullptr, l * P_, P_, NVT, 32, A16);
                gemm(A16, Wgiw, gi_bih, F1, P_, 19, 32, 300, 0);
            } else {
                packA(h, VHS, nullptr, l * P_, P_, VHS, 128, A16);
                gemm(A16, Wghw, gh_bih, F1, P_, 19, 128, 300, 0);
            }
            // GRU hidden projection (hx = aggregated messages)
            packA(PS, 112, nullptr, 0, P_, VHS, 128, A16);
            gemm(A16, (r == 0) ? Wgihh : Wghhh, (r == 0) ? gi_bhh : gh_bhh,
                 F2, P_, 19, 128, 300, 0);
            gru_pointwise<<<P_, 128, 0, stream>>>(F1, 304, F2, PS, 112, nullptr, 0,
                                                  h, VHS, nullptr, l * P_);
        }
        // --- SV update: a = GRU(li, sv_prev, gl), b = GRU(out, a, go) ---
        packA(h, VHS, li_node, 0, S_, VHS, 128, A16);
        gemm(A16, Wglw, gl_bih, F1, S_, 19, 128, 300, 0);
        packA(h, VHS, sv_node, 0, S_, VHS, 128, A16);
        gemm(A16, Wglhh, gl_bhh, F2, S_, 19, 128, 300, 0);
        gru_pointwise<<<S_, 128, 0, stream>>>(F1, 304, F2, h, VHS, sv_node, 0,
                                              ABUF, 112, nullptr, 0);
        outnode_gi<<<1, 320, 0, stream>>>(go_wih, go_bih, h, out_node, GIROW);
        packA(ABUF, 112, nullptr, 0, S_, VHS, 128, A16);
        gemm(A16, Wgohh, go_bhh, F2, S_, 19, 128, 300, 0);
        gru_pointwise<<<S_, 128, 0, stream>>>(GIROW, 0, F2, ABUF, 112, nullptr, 0,
                                              h, VHS, sv_node, 0);
    }

    // --- clause loop: two BiLSTMs + heads ---
    for (int c = 0; c < NCL; c++) {
        if (c == 0) packA(h, VHS, sv_node, 0, S_, VHS, 128, A16);
        else        packA(SVF, 112, nullptr, 0, S_, VHS, 128, A16);
        gemm(A16, Wluf, lu_bih_f, F1, S_, 25, 128, 400, 0);
        gemm(A16, Wlub, lu_bih_b, F2, S_, 25, 128, 400, 0);
        lstm_recur<<<2, 512, 0, stream>>>(F1, F2, lu_whh_f, lu_whh_b,
                                          lu_bhh_f, lu_bhh_b, PROP, 200);
        packA(PROP, 200, nullptr, 0, S_, 200, 224, A16);
        gemm(A16, Wlgf, lg_bih_f, F1, S_, 25, 224, 400, 0);
        gemm(A16, Wlgb, lg_bih_b, F2, S_, 25, 224, 400, 0);
        lstm_recur<<<2, 512, 0, stream>>>(F1, F2, lg_whh_f, lg_whh_b,
                                          lg_bhh_f, lg_bhh_b, PS, 200);   // PS = pred
        if (c + 1 < NCL)  // sv_feature for next clause: prop @ u21^T + b (A16 still holds prop16)
            gemm(A16, Wu21, u21_b, SVF, S_, 7, 224, VHS, 0);
        packA(PS, 200, nullptr, 0, S_, 200, 224, A16);
        gemm(A16, Wm1, m1_b, F1, S_, 2, 224, CHS, 2);                     // relu head
        final_out<<<(S_ + 255) / 256, 256, 0, stream>>>(F1, m2_w, m2_b,
                                                        out + (size_t)c * S_);
    }
}
